// AdvancedHybridGNN_18322330485179
// MI455X (gfx1250) — compile-verified, matrix operand order checked by reading, not run on hardware
//
#include <hip/hip_runtime.h>

#define HID 128
#define IN_DIM 64

typedef __attribute__((ext_vector_type(16))) _Float16 v16h;
typedef __attribute__((ext_vector_type(8)))  float    v8f;

// ---------- float <-> monotone-uint key (for atomic float max) ----------
static __device__ __forceinline__ unsigned f2ord(float f) {
  unsigned u = __float_as_uint(f);
  return (u & 0x80000000u) ? ~u : (u | 0x80000000u);
}
static __device__ __forceinline__ float ord2f(unsigned k) {
  return __uint_as_float((k & 0x80000000u) ? (k ^ 0x80000000u) : ~k);
}

// ---------------------------- fills ----------------------------
__global__ void fill_f32_kernel(float* p, float v, long n) {
  long i = (long)blockIdx.x * blockDim.x + threadIdx.x;
  if (i < n) p[i] = v;
}
__global__ void fill_u32_kernel(unsigned* p, unsigned v, long n) {
  long i = (long)blockIdx.x * blockDim.x + threadIdx.x;
  if (i < n) p[i] = v;
}

// ---------------------------- B pre-pack ----------------------------
// Convert W[Kin x Kout] (fp32, row-major) into per-(coltile, kblock, lane) v16h
// fragments matching the ISA 16-bit B 32x16 layout, zero-padded past Kout.
// Fragment element e of lane (halfw=lane>>4, idx=lane&15):
//   K = kb*32 + halfw*8 + (e<8 ? e : e+8),  N = ct*16 + idx
__global__ void pack_b_kernel(const float* __restrict__ B, _Float16* __restrict__ P,
                              int Kin, int Kout, int coltiles) {
  int i = blockIdx.x * blockDim.x + threadIdx.x;
  int kblocks = Kin >> 5;
  int total = coltiles * kblocks * 32;
  if (i >= total) return;
  int lane = i & 31;
  int rest = i >> 5;
  int kb = rest % kblocks;
  int ct = rest / kblocks;
  int halfw = lane >> 4;
  int idx = lane & 15;
  int col = ct * 16 + idx;
  _Float16* out = P + (size_t)i * 16;
#pragma unroll
  for (int e = 0; e < 16; ++e) {
    int krel = halfw * 8 + ((e < 8) ? e : (e + 8));
    int k = kb * 32 + krel;
    out[e] = (col < Kout) ? (_Float16)B[(size_t)k * Kout + col] : (_Float16)0.f;
  }
}

// ---------------------------- WMMA GEMM ----------------------------
// C[N x KOUT] = act(A[N x KIN] @ W + bias); W pre-packed to f16 fragments.
// One wave computes a 16 x (16*NT) output tile; A fragment (two float4 pairs
// per lane, cvt to f16) is reused across NT v_wmma_f32_16x16x32_f16 issues.
template <int KIN, int KOUT, int ACT, int NT>
__global__ void wmma_gemm_t(const float* __restrict__ A, const _Float16* __restrict__ P,
                            const float* __restrict__ bias, float* __restrict__ C,
                            int total_waves) {
  constexpr int COLT = (KOUT + 15) / 16;
  constexpr int GROUPS = COLT / NT;
  constexpr int KB = KIN / 32;
  int wave = blockIdx.x * (blockDim.x >> 5) + (threadIdx.x >> 5);
  if (wave >= total_waves) return;
  int lane = threadIdx.x & 31;
  int rt = wave / GROUPS;
  int cg = wave - rt * GROUPS;
  int row0 = rt << 4;
  int halfw = lane >> 4;   // K-half select per ISA fragment layout
  int idx = lane & 15;     // row (A) / column (B,C,D) within tile

  v8f acc[NT];
  v8f zero = {0.f, 0.f, 0.f, 0.f, 0.f, 0.f, 0.f, 0.f};
#pragma unroll
  for (int t = 0; t < NT; ++t) acc[t] = zero;

  const float* arow = A + (size_t)(row0 + idx) * KIN + halfw * 8;
#pragma unroll
  for (int kb = 0; kb < KB; ++kb) {
    const float* ap = arow + kb * 32;
    float4 a0 = *(const float4*)(ap);
    float4 a1 = *(const float4*)(ap + 4);
    float4 a2 = *(const float4*)(ap + 16);
    float4 a3 = *(const float4*)(ap + 20);
    v16h af;
    af[0]=(_Float16)a0.x;  af[1]=(_Float16)a0.y;  af[2]=(_Float16)a0.z;  af[3]=(_Float16)a0.w;
    af[4]=(_Float16)a1.x;  af[5]=(_Float16)a1.y;  af[6]=(_Float16)a1.z;  af[7]=(_Float16)a1.w;
    af[8]=(_Float16)a2.x;  af[9]=(_Float16)a2.y;  af[10]=(_Float16)a2.z; af[11]=(_Float16)a2.w;
    af[12]=(_Float16)a3.x; af[13]=(_Float16)a3.y; af[14]=(_Float16)a3.z; af[15]=(_Float16)a3.w;
#pragma unroll
    for (int t = 0; t < NT; ++t) {
      int ct = cg * NT + t;
      v16h bf = *(const v16h*)(P + (((size_t)ct * KB + kb) * 32 + lane) * 16);
      acc[t] = __builtin_amdgcn_wmma_f32_16x16x32_f16(false, af, false, bf,
                                                      (short)0, acc[t], false, false);
    }
  }
#pragma unroll
  for (int t = 0; t < NT; ++t) {
    int coln = (cg * NT + t) * 16 + idx;
    if constexpr ((KOUT & 15) != 0) {
      if (coln >= KOUT) continue;
    }
    float bv = bias ? bias[coln] : 0.f;
#pragma unroll
    for (int r = 0; r < 8; ++r) {   // D layout: lanes 0-15 M=r, lanes 16-31 M=r+8
      float v = acc[t][r] + bv;
      if constexpr (ACT == 1) v = fmaxf(v, 0.f);
      C[(size_t)(row0 + halfw * 8 + r) * KOUT + coln] = v;
    }
  }
}

template <int KIN, int KOUT, int ACT, int NT>
static void launch_gemm_t(const float* A, const float* W, const float* bias, float* C,
                          int N, _Float16* pbuf, hipStream_t s) {
  constexpr int COLT = (KOUT + 15) / 16;
  int packthreads = COLT * (KIN / 32) * 32;
  pack_b_kernel<<<dim3((packthreads + 255) / 256), dim3(256), 0, s>>>(W, pbuf, KIN, KOUT, COLT);
  constexpr int GROUPS = COLT / NT;
  int waves = (N / 16) * GROUPS;
  int blocks = (waves + 7) / 8;   // 256 threads = 8 wave32
  wmma_gemm_t<KIN, KOUT, ACT, NT><<<dim3(blocks), dim3(256), 0, s>>>(A, pbuf, bias, C, waves);
}

// ---------------------------- graph prep ----------------------------
__global__ void deg_kernel(const int* __restrict__ ei, int E, int n, float* __restrict__ deg) {
  int i = blockIdx.x * blockDim.x + threadIdx.x;
  if (i >= E + n) return;
  int d = (i < E) ? ei[E + i] : (i - E);   // self-loops appended
  atomicAdd(&deg[d], 1.0f);
}
__global__ void dis_kernel(float* __restrict__ degdis, int n) {
  int i = blockIdx.x * blockDim.x + threadIdx.x;
  if (i >= n) return;
  float dg = degdis[i];
  degdis[i] = (dg > 0.f) ? rsqrtf(fmaxf(dg, 1.f)) : 0.f;
}
__global__ void norm_kernel(const int* __restrict__ ei, int E, int n,
                            const float* __restrict__ dis, float* __restrict__ nrm) {
  int i = blockIdx.x * blockDim.x + threadIdx.x;
  if (i >= E + n) return;
  int s = (i < E) ? ei[i]     : (i - E);
  int d = (i < E) ? ei[E + i] : (i - E);
  nrm[i] = dis[s] * dis[d];
}

// ---------------------------- GCN scatter ----------------------------
// 32 lanes per edge, float4 per lane -> 128 features; atomic-add into L2-resident table
__global__ void gcn_scatter_kernel(const int* __restrict__ ei, int E, int n,
                                   const float* __restrict__ t, const float* __restrict__ nrm,
                                   float* __restrict__ acc) {
  long tid = (long)blockIdx.x * blockDim.x + threadIdx.x;
  if (tid >= (long)(E + n) * 32) return;
  int edge = (int)(tid >> 5);
  int f = ((int)tid & 31) * 4;
  int s = (edge < E) ? ei[edge]     : (edge - E);
  int d = (edge < E) ? ei[E + edge] : (edge - E);
  float w = nrm[edge];
  float4 v = *(const float4*)(t + (size_t)s * HID + f);
  float* o = acc + (size_t)d * HID + f;
  atomicAdd(o + 0, v.x * w);
  atomicAdd(o + 1, v.y * w);
  atomicAdd(o + 2, v.z * w);
  atomicAdd(o + 3, v.w * w);
}

// ---------------------------- elementwise post ----------------------------
// v = x (+bias); bnmode 1: v = v*(g/sqrt(1+eps)) + bb; bnmode 2: v *= g
// act 1: relu, act 2: elu; optional residual add; optional residual copy-out
__global__ void postproc_kernel(const float* __restrict__ x, const float* __restrict__ bias,
                                const float* __restrict__ g, const float* __restrict__ bb,
                                const float* __restrict__ res_in, float* __restrict__ out,
                                float* __restrict__ res_out, long total, int C,
                                int bnmode, int act) {
  long i = (long)blockIdx.x * blockDim.x + threadIdx.x;
  if (i >= total) return;
  int c = (int)(i % C);
  float v = x[i];
  if (bias) v += bias[c];
  if (bnmode == 1) v = v * (g[c] * rsqrtf(1.0f + 1e-5f)) + bb[c];
  else if (bnmode == 2) v = v * g[c];
  if (act == 1) v = fmaxf(v, 0.f);
  else if (act == 2) v = (v > 0.f) ? v : (__expf(v) - 1.f);
  if (res_in) v += res_in[i];
  out[i] = v;
  if (res_out) res_out[i] = v;
}

__global__ void fuse_kernel(const float* __restrict__ a, const float* __restrict__ b,
                            float* __restrict__ o, long total) {
  long i = (long)blockIdx.x * blockDim.x + threadIdx.x;
  if (i < total) o[i] = 0.6f * a[i] + 0.4f * b[i];
}

// ---------------------------- GAT kernels ----------------------------
__global__ void gat_node_attn_kernel(const float* __restrict__ xh, const float* __restrict__ aS,
                                     const float* __restrict__ aD, float* __restrict__ asrc,
                                     float* __restrict__ adst, int n, int heads, int C) {
  int i = blockIdx.x * blockDim.x + threadIdx.x;
  if (i >= n * heads) return;
  int node = i / heads, h = i - node * heads;
  const float* xr = xh + (size_t)node * HID + h * C;
  const float* as = aS + h * C;
  const float* ad = aD + h * C;
  float s = 0.f, d = 0.f;
  for (int c = 0; c < C; ++c) { float v = xr[c]; s += v * as[c]; d += v * ad[c]; }
  asrc[i] = s; adst[i] = d;
}

__global__ void gat_edge_max_kernel(const int* __restrict__ ei, int E, int n, int heads,
                                    const float* __restrict__ asrc, const float* __restrict__ adst,
                                    float* __restrict__ ev, unsigned* __restrict__ segmax) {
  long i = (long)blockIdx.x * blockDim.x + threadIdx.x;
  if (i >= (long)(E + n) * heads) return;
  int edge = (int)(i / heads), h = (int)(i - (long)edge * heads);
  int s = (edge < E) ? ei[edge]     : (edge - E);
  int d = (edge < E) ? ei[E + edge] : (edge - E);
  float e = asrc[s * heads + h] + adst[d * heads + h];
  e = (e > 0.f) ? e : 0.2f * e;                     // leaky_relu(0.2)
  ev[i] = e;
  atomicMax(&segmax[d * heads + h], f2ord(e));
}

__global__ void gat_edge_sum_kernel(const int* __restrict__ ei, int E, int n, int heads,
                                    const unsigned* __restrict__ segmax, float* __restrict__ ev,
                                    float* __restrict__ segsum) {
  long i = (long)blockIdx.x * blockDim.x + threadIdx.x;
  if (i >= (long)(E + n) * heads) return;
  int edge = (int)(i / heads), h = (int)(i - (long)edge * heads);
  int d = (edge < E) ? ei[E + edge] : (edge - E);
  float m = ord2f(segmax[d * heads + h]);
  float ex = __expf(ev[i] - m);
  ev[i] = ex;                                       // keep numerator for scatter pass
  atomicAdd(&segsum[d * heads + h], ex);
}

__global__ void gat_scatter_kernel(const int* __restrict__ ei, int E, int n, int heads, int C,
                                   const float* __restrict__ xh, const float* __restrict__ ev,
                                   const float* __restrict__ segsum, float* __restrict__ acc) {
  long tid = (long)blockIdx.x * blockDim.x + threadIdx.x;
  if (tid >= (long)(E + n) * 32) return;
  int edge = (int)(tid >> 5);
  int f = ((int)tid & 31) * 4;
  int h = f / C;
  int s = (edge < E) ? ei[edge]     : (edge - E);
  int d = (edge < E) ? ei[E + edge] : (edge - E);
  float alpha = ev[(size_t)edge * heads + h] / (segsum[d * heads + h] + 1e-16f);
  float4 v = *(const float4*)(xh + (size_t)s * HID + f);
  float* o = acc + (size_t)d * HID + f;
  atomicAdd(o + 0, v.x * alpha);
  atomicAdd(o + 1, v.y * alpha);
  atomicAdd(o + 2, v.z * alpha);
  atomicAdd(o + 3, v.w * alpha);
}

// ---------------------------- launch ----------------------------
extern "C" void kernel_launch(void* const* d_in, const int* in_sizes, int n_in,
                              void* d_out, int out_size, void* d_ws, size_t ws_size,
                              hipStream_t stream) {
  (void)n_in; (void)out_size; (void)ws_size;
  const int N = in_sizes[0] / IN_DIM;
  const int E = in_sizes[1] / 2;
  const int Etot = E + N;

  const float* x        = (const float*)d_in[0];
  const int*   ei       = (const int*)  d_in[1];
  const float* gcn_in_w = (const float*)d_in[2];
  const float* gcn_in_b = (const float*)d_in[3];
  const float* gcn_res_w= (const float*)d_in[4];
  const float* gcn_res_b= (const float*)d_in[5];
  const float* gcn_w    = (const float*)d_in[6];
  const float* gcn_b    = (const float*)d_in[7];
  const float* gcn_bn_g = (const float*)d_in[8];
  const float* gcn_bn_b = (const float*)d_in[9];
  const float* gcn_c1w  = (const float*)d_in[10];
  const float* gcn_c1b  = (const float*)d_in[11];
  const float* gcn_c2w  = (const float*)d_in[12];
  const float* gcn_c2b  = (const float*)d_in[13];
  const float* gcn_c3w  = (const float*)d_in[14];
  const float* gcn_c3b  = (const float*)d_in[15];
  const float* gat_in_w = (const float*)d_in[16];
  const float* gat_in_b = (const float*)d_in[17];
  const float* gat_w    = (const float*)d_in[18];
  const float* gat_asrc = (const float*)d_in[19];
  const float* gat_adst = (const float*)d_in[20];
  const float* gat_b    = (const float*)d_in[21];
  const float* gat_bn_g = (const float*)d_in[22];
  const float* gat_bn_b = (const float*)d_in[23];
  const float* feat_imp = (const float*)d_in[24];
  const float* gat_c1w  = (const float*)d_in[25];
  const float* gat_c1b  = (const float*)d_in[26];
  const float* gat_cbn_g= (const float*)d_in[27];
  const float* gat_cbn_b= (const float*)d_in[28];
  const float* gat_c2w  = (const float*)d_in[29];
  const float* gat_c2b  = (const float*)d_in[30];
  const float* gat_c3w  = (const float*)d_in[31];
  const float* gat_c3b  = (const float*)d_in[32];
  const float* fin_w    = (const float*)d_in[33];
  const float* fin_b    = (const float*)d_in[34];
  const float* fin_bn_g = (const float*)d_in[35];
  const float* fin_bn_b = (const float*)d_in[36];
  const float* fin2_w   = (const float*)d_in[37];
  const float* fin2_b   = (const float*)d_in[38];

  // ---- workspace carve ----
  char* wp = (char*)d_ws;
  auto take = [&](size_t bytes) -> void* {
    void* r = (void*)wp;
    wp += (bytes + 255) & ~(size_t)255;
    return r;
  };
  float*    degdis  = (float*)   take((size_t)N * 4);
  float*    nrm     = (float*)   take((size_t)Etot * 4);
  float*    xp      = (float*)   take((size_t)N * HID * 4);   // GCN act / later GAT act (xg)
  float*    res     = (float*)   take((size_t)N * HID * 4);   // GCN residual
  float*    tbuf    = (float*)   take((size_t)N * HID * 4);   // per-layer x@W
  float*    acc     = (float*)   take((size_t)N * HID * 4);   // aggregation accumulator / fused
  float*    gcn_out = (float*)   take((size_t)N * HID * 4);
  float*    gat_out = (float*)   take((size_t)N * HID * 4);
  float*    h64     = (float*)   take((size_t)N * 64 * 4);
  float*    h32     = (float*)   take((size_t)N * 32 * 4);
  float*    asrcb   = (float*)   take((size_t)N * 4 * 4);
  float*    adstb   = (float*)   take((size_t)N * 4 * 4);
  unsigned* segmax  = (unsigned*)take((size_t)N * 4 * 4);
  float*    segsum  = (float*)   take((size_t)N * 4 * 4);
  float*    evbuf   = (float*)   take((size_t)Etot * 4 * 4);
  _Float16* pbuf    = (_Float16*)take((size_t)64 * 1024);     // packed weight fragments

  dim3 B256(256);
  auto gb = [](long t) { return dim3((unsigned)((t + 255) / 256)); };
  auto post = [&](const float* xin, const float* bias, const float* g, const float* bb,
                  const float* rin, float* outp, float* rout, int C, int bnmode, int act) {
    long tot = (long)N * C;
    postproc_kernel<<<gb(tot), B256, 0, stream>>>(xin, bias, g, bb, rin, outp, rout, tot, C, bnmode, act);
  };

  // ---- graph normalization ----
  fill_f32_kernel<<<gb(N), B256, 0, stream>>>(degdis, 0.f, N);
  deg_kernel<<<gb(Etot), B256, 0, stream>>>(ei, E, N, degdis);
  dis_kernel<<<gb(N), B256, 0, stream>>>(degdis, N);
  norm_kernel<<<gb(Etot), B256, 0, stream>>>(ei, E, N, degdis, nrm);

  // ---- GCN branch ----
  launch_gemm_t<IN_DIM, HID, 0, 2>(x, gcn_in_w, gcn_in_b, xp, N, pbuf, stream);
  launch_gemm_t<IN_DIM, HID, 0, 2>(x, gcn_res_w, gcn_res_b, res, N, pbuf, stream);
  for (int i = 0; i < 3; ++i) {
    launch_gemm_t<HID, HID, 0, 2>(xp, gcn_w + (size_t)i * HID * HID, nullptr, tbuf, N, pbuf, stream);
    fill_f32_kernel<<<gb((long)N * HID), B256, 0, stream>>>(acc, 0.f, (long)N * HID);
    gcn_scatter_kernel<<<gb((long)Etot * 32), B256, 0, stream>>>(ei, E, N, tbuf, nrm, acc);
    if (i == 1)
      post(acc, gcn_b + i * HID, gcn_bn_g + i * HID, gcn_bn_b + i * HID, res, xp, res, HID, 1, 1);
    else
      post(acc, gcn_b + i * HID, gcn_bn_g + i * HID, gcn_bn_b + i * HID, nullptr, xp, nullptr, HID, 1, 1);
  }
  launch_gemm_t<HID, 64, 1, 2>(xp, gcn_c1w, gcn_c1b, h64, N, pbuf, stream);
  launch_gemm_t<64, 32, 1, 2>(h64, gcn_c2w, gcn_c2b, h32, N, pbuf, stream);
  launch_gemm_t<32, HID, 0, 2>(h32, gcn_c3w, gcn_c3b, gcn_out, N, pbuf, stream);

  // ---- GAT branch (xg lives in xp) ----
  launch_gemm_t<IN_DIM, HID, 0, 2>(x, gat_in_w, gat_in_b, xp, N, pbuf, stream);
  const int heads_arr[3] = {4, 4, 1};
  unsigned negkey;
  { union { float f; unsigned u; } c; c.f = -1e30f;
    negkey = (c.u & 0x80000000u) ? ~c.u : (c.u | 0x80000000u); }
  for (int i = 0; i < 3; ++i) {
    int heads = heads_arr[i];
    int C = HID / heads;
    launch_gemm_t<HID, HID, 0, 2>(xp, gat_w + (size_t)i * HID * HID, nullptr, tbuf, N, pbuf, stream);
    gat_node_attn_kernel<<<gb((long)N * heads), B256, 0, stream>>>(
        tbuf, gat_asrc + i * HID, gat_adst + i * HID, asrcb, adstb, N, heads, C);
    fill_u32_kernel<<<gb((long)N * heads), B256, 0, stream>>>(segmax, negkey, (long)N * heads);
    fill_f32_kernel<<<gb((long)N * heads), B256, 0, stream>>>(segsum, 0.f, (long)N * heads);
    gat_edge_max_kernel<<<gb((long)Etot * heads), B256, 0, stream>>>(
        ei, E, N, heads, asrcb, adstb, evbuf, segmax);
    gat_edge_sum_kernel<<<gb((long)Etot * heads), B256, 0, stream>>>(
        ei, E, N, heads, segmax, evbuf, segsum);
    fill_f32_kernel<<<gb((long)N * HID), B256, 0, stream>>>(acc, 0.f, (long)N * HID);
    gat_scatter_kernel<<<gb((long)Etot * 32), B256, 0, stream>>>(
        ei, E, N, heads, C, tbuf, evbuf, segsum, acc);
    post(acc, gat_b + i * HID, gat_bn_g + i * HID, gat_bn_b + i * HID, nullptr, xp, nullptr, HID, 1, 2);
  }
  post(xp, nullptr, feat_imp, nullptr, nullptr, xp, nullptr, HID, 2, 0);   // xg *= feat_imp
  launch_gemm_t<HID, 64, 1, 2>(xp, gat_c1w, gat_c1b, h64, N, pbuf, stream);
  post(h64, nullptr, gat_cbn_g, gat_cbn_b, nullptr, h64, nullptr, 64, 1, 0);
  launch_gemm_t<64, 32, 1, 2>(h64, gat_c2w, gat_c2b, h32, N, pbuf, stream);
  launch_gemm_t<32, HID, 0, 2>(h32, gat_c3w, gat_c3b, gat_out, N, pbuf, stream);

  // ---- fusion head ----
  fuse_kernel<<<gb((long)N * HID), B256, 0, stream>>>(gcn_out, gat_out, acc, (long)N * HID);
  launch_gemm_t<HID, 64, 1, 2>(acc, fin_w, fin_b, h64, N, pbuf, stream);
  post(h64, nullptr, fin_bn_g, fin_bn_b, nullptr, h64, nullptr, 64, 1, 0);
  launch_gemm_t<64, 2, 0, 1>(h64, fin2_w, fin2_b, (float*)d_out, N, pbuf, stream);
}
